// ConcreteAttentionModel_47717086659102
// MI455X (gfx1250) — matrix operand order repaired and checked by reading, not compiled, not run on hardware
//
#include <hip/hip_runtime.h>

// ---------------------------------------------------------------------------
// CDNA5 / gfx1250 WMMA helpers (wave32)
// ---------------------------------------------------------------------------
typedef __attribute__((ext_vector_type(16))) __bf16 bf16x16;
typedef __attribute__((ext_vector_type(8)))  float  v8f;
typedef __attribute__((ext_vector_type(4)))  float  f32x4;

__device__ inline unsigned short f2bf(float f) {
  unsigned u = __builtin_bit_cast(unsigned, f);
  unsigned r = u + 0x7FFFu + ((u >> 16) & 1u);   // round-to-nearest-even
  return (unsigned short)(r >> 16);
}

// Packed fp32x2 -> bf16x2 in one dword (low = first arg).
// Preference: v_cvt_pk_bf16_f32 > v_perm_b32 round-to-nearest > manual RNE.
#if __has_builtin(__builtin_amdgcn_cvt_pk_bf16_f32)
typedef __attribute__((ext_vector_type(2))) __bf16 bf16x2;
__device__ inline unsigned cvt2(float lo, float hi) {
  bf16x2 r = __builtin_amdgcn_cvt_pk_bf16_f32(lo, hi);
  return __builtin_bit_cast(unsigned, r);
}
#elif __has_builtin(__builtin_amdgcn_perm)
__device__ inline unsigned cvt2(float lo, float hi) {
  unsigned ul = __builtin_bit_cast(unsigned, lo) + 0x8000u;  // round-to-nearest
  unsigned uh = __builtin_bit_cast(unsigned, hi) + 0x8000u;
  // dst bytes = {ul.b2, ul.b3, uh.b2, uh.b3}; {src0=uh(hi dword), src1=ul(lo dword)}
  return __builtin_amdgcn_perm(uh, ul, 0x07060302u);
}
#else
__device__ inline unsigned cvt2(float lo, float hi) {
  return (unsigned)f2bf(lo) | ((unsigned)f2bf(hi) << 16);
}
#endif

union BfFrag { unsigned w[8]; bf16x16 v; };

__device__ inline bf16x16 pack16(f32x4 x0, f32x4 x1, f32x4 x2, f32x4 x3) {
  BfFrag f;
  f.w[0] = cvt2(x0[0], x0[1]); f.w[1] = cvt2(x0[2], x0[3]);
  f.w[2] = cvt2(x1[0], x1[1]); f.w[3] = cvt2(x1[2], x1[3]);
  f.w[4] = cvt2(x2[0], x2[1]); f.w[5] = cvt2(x2[2], x2[3]);
  f.w[6] = cvt2(x3[0], x3[1]); f.w[7] = cvt2(x3[2], x3[3]);
  return f.v;
}

__device__ inline v8f wmma_bf16(bf16x16 a, bf16x16 b, v8f c) {
  return __builtin_amdgcn_wmma_f32_16x16x32_bf16(
      /*neg_a=*/false, a, /*neg_b=*/false, b,
      /*c_mod=*/(short)0, c, /*reuse_a=*/false, /*reuse_b=*/false);
}

// A fragment: 16x32 tile, rows rbase.., fp32 source (stride ld).
// Lane layout: m=lane&15, half=lane>>4; K = half*16 + e.
__device__ inline bf16x16 load_a_frag(const float* __restrict__ A, long ld,
                                      int rbase, int k0, int lane) {
  int half = lane >> 4, lm = lane & 15;
  const f32x4* q = (const f32x4*)(A + (long)(rbase + lm) * ld + k0 + half * 16);
  return pack16(q[0], q[1], q[2], q[3]);
}

// Same, but non-temporal (streaming weights with zero reuse: Wz1, Wpf).
__device__ inline bf16x16 load_a_frag_nt(const float* __restrict__ A, long ld,
                                         int rbase, int k0, int lane) {
  int half = lane >> 4, lm = lane & 15;
  const f32x4* q = (const f32x4*)(A + (long)(rbase + lm) * ld + k0 + half * 16);
  f32x4 x0 = __builtin_nontemporal_load(q);
  f32x4 x1 = __builtin_nontemporal_load(q + 1);
  f32x4 x2 = __builtin_nontemporal_load(q + 2);
  f32x4 x3 = __builtin_nontemporal_load(q + 3);
  return pack16(x0, x1, x2, x3);
}

// B fragment: B[k,n] = W[nbase+n][k], fp32 row-major W (stride ld).
// Lane layout: n=lane&15, khalf=lane>>4; element e <-> K = khalf*16 + e.
__device__ inline bf16x16 load_b_frag_w(const float* __restrict__ W, long ld,
                                        int nbase, int k0, int lane) {
  int half = lane >> 4, lm = lane & 15;
  const f32x4* q = (const f32x4*)(W + (long)(nbase + lm) * ld + k0 + half * 16);
  return pack16(q[0], q[1], q[2], q[3]);
}

// B fragment from pre-converted bf16 row-major matrix (rows = N dim).
__device__ inline bf16x16 load_b_frag_bf(const unsigned short* __restrict__ P, long ld,
                                         int nbase, int k0, int lane) {
  int half = lane >> 4, lm = lane & 15;
  return *(const bf16x16*)(P + (long)(nbase + lm) * ld + k0 + half * 16);  // 32B aligned
}

// ---------------------------------------------------------------------------
// fp32 -> bf16 matrix pre-conversion (vectorized, n multiple of 4)
// ---------------------------------------------------------------------------
__global__ void k_cvt_bf16(const float* __restrict__ x, unsigned short* __restrict__ y,
                           long n) {
  long i = ((long)blockIdx.x * 256 + threadIdx.x) * 4;
  if (i >= n) return;
  f32x4 v = *(const f32x4*)(x + i);
  unsigned* py = (unsigned*)(y + i);
  py[0] = cvt2(v[0], v[1]);
  py[1] = cvt2(v[2], v[3]);
}

// ---------------------------------------------------------------------------
// Stage 1: img = relu(image_features @ W_ic^T + b_ic)   [1024,65536]x[65536,512]
// grid(16,8) block 256. Wave = 16M x 32N, 8 waves = 4M x 2N -> WG tile 64x64.
// Two variants: B from pre-converted bf16 (preferred) or from fp32 (fallback).
// ---------------------------------------------------------------------------
__global__ __launch_bounds__(256) void k_img_gemm_bf(const float* __restrict__ A,
                                                     const unsigned short* __restrict__ Wb16,
                                                     const float* __restrict__ bias,
                                                     float* __restrict__ out) {
  const long F = 65536;
  int lane = threadIdx.x & 31, wave = threadIdx.x >> 5;
  int wm = wave & 3, wn = wave >> 2;
  int mbase = blockIdx.x * 64 + wm * 16;
  int nbase = blockIdx.y * 64 + wn * 32;
  v8f acc0 = {}, acc1 = {};
  for (int k0 = 0; k0 < 65536; k0 += 32) {
    bf16x16 a  = load_a_frag(A, F, mbase, k0, lane);
    bf16x16 b0 = load_b_frag_bf(Wb16, F, nbase, k0, lane);
    bf16x16 b1 = load_b_frag_bf(Wb16, F, nbase + 16, k0, lane);
    acc0 = wmma_bf16(a, b0, acc0);
    acc1 = wmma_bf16(a, b1, acc1);
  }
  int half = lane >> 4, lm = lane & 15;
#pragma unroll
  for (int v = 0; v < 8; ++v) {
    int m = mbase + v + 8 * half;
    int c0 = nbase + lm, c1 = nbase + 16 + lm;
    out[(long)m * 512 + c0] = fmaxf(acc0[v] + bias[c0], 0.f);
    out[(long)m * 512 + c1] = fmaxf(acc1[v] + bias[c1], 0.f);
  }
}

__global__ __launch_bounds__(256) void k_img_gemm(const float* __restrict__ A,
                                                  const float* __restrict__ W,
                                                  const float* __restrict__ bias,
                                                  float* __restrict__ out) {
  const long F = 65536;
  int lane = threadIdx.x & 31, wave = threadIdx.x >> 5;
  int wm = wave & 3, wn = wave >> 2;
  int mbase = blockIdx.x * 64 + wm * 16;
  int nbase = blockIdx.y * 64 + wn * 32;
  v8f acc0 = {}, acc1 = {};
  for (int k0 = 0; k0 < 65536; k0 += 32) {
    bf16x16 a  = load_a_frag(A, F, mbase, k0, lane);
    bf16x16 b0 = load_b_frag_w(W, F, nbase, k0, lane);
    bf16x16 b1 = load_b_frag_w(W, F, nbase + 16, k0, lane);
    acc0 = wmma_bf16(a, b0, acc0);
    acc1 = wmma_bf16(a, b1, acc1);
  }
  int half = lane >> 4, lm = lane & 15;
#pragma unroll
  for (int v = 0; v < 8; ++v) {
    int m = mbase + v + 8 * half;
    int c0 = nbase + lm, c1 = nbase + 16 + lm;
    out[(long)m * 512 + c0] = fmaxf(acc0[v] + bias[c0], 0.f);
    out[(long)m * 512 + c1] = fmaxf(acc1[v] + bias[c1], 0.f);
  }
}

// ---------------------------------------------------------------------------
// Stage 2: h = tanh(img@Wa^T+ba) * sigmoid(img@Wb^T+bb)  -> [1024,256]
// Wa/Wb pre-converted to bf16. grid(16,4) block 256.
// ---------------------------------------------------------------------------
__global__ __launch_bounds__(256) void k_attn_gemm(const float* __restrict__ img,
                                                   const unsigned short* __restrict__ WaB,
                                                   const float* __restrict__ ba,
                                                   const unsigned short* __restrict__ WbB,
                                                   const float* __restrict__ bb,
                                                   float* __restrict__ h) {
  int lane = threadIdx.x & 31, wave = threadIdx.x >> 5;
  int wm = wave & 3, wn = wave >> 2;
  int mbase = blockIdx.x * 64 + wm * 16;
  int nbase = blockIdx.y * 64 + wn * 32;
  v8f aa0 = {}, aa1 = {}, gg0 = {}, gg1 = {};
  for (int k0 = 0; k0 < 512; k0 += 32) {
    bf16x16 a   = load_a_frag(img, 512, mbase, k0, lane);
    bf16x16 ba0 = load_b_frag_bf(WaB, 512, nbase, k0, lane);
    bf16x16 ba1 = load_b_frag_bf(WaB, 512, nbase + 16, k0, lane);
    bf16x16 bb0 = load_b_frag_bf(WbB, 512, nbase, k0, lane);
    bf16x16 bb1 = load_b_frag_bf(WbB, 512, nbase + 16, k0, lane);
    aa0 = wmma_bf16(a, ba0, aa0);
    aa1 = wmma_bf16(a, ba1, aa1);
    gg0 = wmma_bf16(a, bb0, gg0);
    gg1 = wmma_bf16(a, bb1, gg1);
  }
  int half = lane >> 4, lm = lane & 15;
#pragma unroll
  for (int v = 0; v < 8; ++v) {
    int m = mbase + v + 8 * half;
    int c0 = nbase + lm, c1 = nbase + 16 + lm;
    float t0 = tanhf(aa0[v] + ba[c0]);
    float t1 = tanhf(aa1[v] + ba[c1]);
    float s0 = 1.f / (1.f + __expf(-(gg0[v] + bb[c0])));
    float s1 = 1.f / (1.f + __expf(-(gg1[v] + bb[c1])));
    h[m * 256 + c0] = t0 * s0;
    h[m * 256 + c1] = t1 * s1;
  }
}

// ---------------------------------------------------------------------------
// Small scalar stages
// ---------------------------------------------------------------------------
__global__ void k_zero(float* p, int n) {
  int i = blockIdx.x * 256 + threadIdx.x;
  if (i < n) p[i] = 0.f;
}

__global__ void k_score(const float* __restrict__ h, const float* __restrict__ Wc,
                        const float* __restrict__ bc, float* __restrict__ s) {
  int m = blockIdx.x * 256 + threadIdx.x;
  if (m >= 1024) return;
  float acc = bc[0];
  for (int c = 0; c < 256; ++c) acc += h[m * 256 + c] * Wc[c];
  s[m] = acc;
}

__global__ __launch_bounds__(256) void k_pool(const float* __restrict__ img,
                                              const float* __restrict__ score,
                                              float* __restrict__ pool) {
  int b = blockIdx.x;
  __shared__ float w[64];
  __shared__ float invs;
  if (threadIdx.x == 0) {
    float mx = -1e30f;
    for (int n = 0; n < 64; ++n) mx = fmaxf(mx, score[b * 64 + n]);
    float sum = 0.f;
    for (int n = 0; n < 64; ++n) { float e = __expf(score[b * 64 + n] - mx); w[n] = e; sum += e; }
    invs = 1.f / (sum * 64.f);     // softmax combined with mean over N
  }
  __syncthreads();
  float inv = invs;
  for (int d = threadIdx.x; d < 512; d += 256) {
    float acc = 0.f;
    for (int n = 0; n < 64; ++n) acc += w[n] * img[((long)b * 64 + n) * 512 + d];
    pool[b * 512 + d] = acc * inv;
  }
}

// text branch: relu(text @ Wt^T + bt); also emits bf16 copy for WMMA B use.
__global__ void k_text(const float* __restrict__ tf, const float* __restrict__ Wt,
                       const float* __restrict__ bt, float* __restrict__ outr,
                       unsigned short* __restrict__ outb) {
  int idx = blockIdx.x * 256 + threadIdx.x;
  if (idx >= 8192) return;
  int b = idx >> 9, d = idx & 511;
  float acc = bt[d];
  for (int t = 0; t < 18; ++t) acc += tf[b * 18 + t] * Wt[d * 18 + t];
  float r = fmaxf(acc, 0.f);
  outr[idx] = r;
  outb[idx] = f2bf(r);
}

__global__ void k_dense512_relu(const float* __restrict__ x, const float* __restrict__ W,
                                const float* __restrict__ bias, float* __restrict__ y) {
  int idx = blockIdx.x * 256 + threadIdx.x;
  if (idx >= 8192) return;
  int b = idx >> 9, d = idx & 511;
  const float* xr = x + b * 512;
  const float* wr = W + d * 512;
  float acc = bias[d];
  for (int k = 0; k < 512; ++k) acc += xr[k] * wr[k];
  y[idx] = fmaxf(acc, 0.f);
}

// ---------------------------------------------------------------------------
// Stage 4: z1[b,o] = sum_i pool[b,i] * (sum_j Wz1[o*512+i, j] * text[b,j])
// WMMA GEMM M=262144 K=512 N=16; i-reduction folded in via f32 atomics.
// Wz1 streamed non-temporally (536 MB, zero reuse). grid(2048) block 256.
// ---------------------------------------------------------------------------
__global__ __launch_bounds__(256) void k_z1(const float* __restrict__ Wz1,
                                            const unsigned short* __restrict__ textB,
                                            const float* __restrict__ pool,
                                            float* __restrict__ z1) {
  int lane = threadIdx.x & 31, wave = threadIdx.x >> 5;
  int tile = blockIdx.x * 8 + wave;      // 16384 tiles
  int rbase = tile * 16;                 // row r = o*512 + i
  int o = rbase >> 9;
  int ibase = rbase & 511;
  v8f acc = {};
  for (int k0 = 0; k0 < 512; k0 += 32) {
    bf16x16 a = load_a_frag_nt(Wz1, 512, rbase, k0, lane);
    bf16x16 b = load_b_frag_bf(textB, 512, 0, k0, lane);  // rows = batches
    acc = wmma_bf16(a, b, acc);
  }
  int half = lane >> 4, lm = lane & 15;  // lm = batch b
  float s = 0.f;
#pragma unroll
  for (int v = 0; v < 8; ++v) {
    int i = ibase + v + 8 * half;
    s += pool[lm * 512 + i] * acc[v];
  }
  atomicAdd(&z1[lm * 512 + o], s);
}

__global__ void k_gate(const float* __restrict__ z1, const float* __restrict__ bz1,
                       const float* __restrict__ h1, float* __restrict__ gz) {
  int idx = blockIdx.x * 256 + threadIdx.x;
  if (idx >= 8192) return;
  int d = idx & 511;
  float z = z1[idx] + bz1[d];
  gz[idx] = h1[idx] / (1.f + __expf(-z));
}

// Kronecker matrix P[b][k] = o1[b, k>>9] * text[b, k&511]  (bf16, [16][262144])
// 2 elements per thread through packed cvt2.
__global__ void k_pmat(const float* __restrict__ o1, const float* __restrict__ textr,
                       unsigned* __restrict__ P32) {
  int t = blockIdx.x * 256 + threadIdx.x;       // 2,097,152 dword slots
  int b = t >> 17;
  int k = (t & 131071) * 2;                     // even k; pair shares i
  int i = k >> 9, j = k & 511;
  float o = o1[b * 512 + i];
  const float* tr = textr + b * 512 + j;
  P32[t] = cvt2(o * tr[0], o * tr[1]);
}

// ---------------------------------------------------------------------------
// Stage 5: fused_acc[d,b] = sum_k Wpf[d,k] * P[b,k]. M=512 K=262144 N=16,
// split-K WMMA, f32 atomic reduction. Wpf streamed non-temporally.
// grid(32,16) block 256; each wave owns a 2048-wide K chunk (64 iters).
// ---------------------------------------------------------------------------
__global__ __launch_bounds__(256) void k_fused(const float* __restrict__ Wpf,
                                               const unsigned short* __restrict__ P,
                                               float* __restrict__ fused) {
  int lane = threadIdx.x & 31, wave = threadIdx.x >> 5;
  int dbase = blockIdx.x * 16;
  int kc = blockIdx.y * 8 + wave;        // 128 K chunks
  int kstart = kc * 2048;
  v8f acc = {};
  for (int k0 = kstart; k0 < kstart + 2048; k0 += 32) {
    bf16x16 a = load_a_frag_nt(Wpf, 262144, dbase, k0, lane);
    bf16x16 b = load_b_frag_bf(P, 262144, 0, k0, lane);
    acc = wmma_bf16(a, b, acc);
  }
  int half = lane >> 4, lm = lane & 15;
#pragma unroll
  for (int v = 0; v < 8; ++v)
    atomicAdd(&fused[(dbase + v + 8 * half) * 16 + lm], acc[v]);
}

__global__ __launch_bounds__(256) void k_final(const float* __restrict__ fused,
                                               const float* __restrict__ bpf,
                                               const float* __restrict__ Wcls,
                                               const float* __restrict__ bcls,
                                               float* __restrict__ out) {
  int b = blockIdx.x;
  __shared__ float red[256];
  float acc = 0.f;
  for (int d = threadIdx.x; d < 512; d += 256)
    acc += fmaxf(fused[d * 16 + b] + bpf[d], 0.f) * Wcls[d];
  red[threadIdx.x] = acc;
  __syncthreads();
  for (int s = 128; s > 0; s >>= 1) {
    if (threadIdx.x < s) red[threadIdx.x] += red[threadIdx.x + s];
    __syncthreads();
  }
  if (threadIdx.x == 0) out[b] = red[0] + bcls[0];
}

// ---------------------------------------------------------------------------
extern "C" void kernel_launch(void* const* d_in, const int* in_sizes, int n_in,
                              void* d_out, int out_size, void* d_ws, size_t ws_size,
                              hipStream_t stream) {
  const float* imgF  = (const float*)d_in[0];
  const float* textF = (const float*)d_in[1];
  const float* W_ic  = (const float*)d_in[2];
  const float* b_ic  = (const float*)d_in[3];
  const float* Wa    = (const float*)d_in[4];
  const float* ba    = (const float*)d_in[5];
  const float* Wb    = (const float*)d_in[6];
  const float* bb    = (const float*)d_in[7];
  const float* Wc    = (const float*)d_in[8];
  const float* bc    = (const float*)d_in[9];
  const float* Wt    = (const float*)d_in[10];
  const float* bt    = (const float*)d_in[11];
  const float* Wh1   = (const float*)d_in[12];
  const float* bh1   = (const float*)d_in[13];
  const float* Wz1   = (const float*)d_in[14];
  const float* bz1   = (const float*)d_in[15];
  const float* Wo1   = (const float*)d_in[16];
  const float* bo1   = (const float*)d_in[17];
  const float* Wpf   = (const float*)d_in[18];
  const float* bpf   = (const float*)d_in[19];
  const float* Wcls  = (const float*)d_in[20];
  const float* bcls  = (const float*)d_in[21];
  float* out = (float*)d_out;

  float* ws = (float*)d_ws;
  float* img   = ws + 0;        // [1024,512] f32
  float* h     = ws + 524288;   // [1024,256] f32
  float* score = ws + 786432;   // [1024]
  float* pool  = ws + 787456;   // [16,512]
  float* textr = ws + 795648;   // [16,512]
  float* h1    = ws + 803840;   // [16,512]
  float* z1    = ws + 812032;   // [16,512]
  float* gz    = ws + 820224;   // [16,512]
  float* o1    = ws + 828416;   // [16,512]
  float* fused = ws + 836608;   // [512,16]
  unsigned short* P      = (unsigned short*)(ws + 844800);   // [16][262144] bf16 (8 MB)
  unsigned short* WaB    = (unsigned short*)(ws + 2941952);  // [256][512]  bf16
  unsigned short* WbB    = (unsigned short*)(ws + 3007488);  // [256][512]  bf16
  unsigned short* textB  = (unsigned short*)(ws + 3073024);  // [16][512]   bf16
  unsigned short* WicB   = (unsigned short*)(ws + 3077120);  // [512][65536] bf16 (64 MB)
  const size_t WS_NEED_WICB = (size_t)3077120 * 4 + (size_t)33554432 * 2;
  bool preconv_wic = ws_size >= WS_NEED_WICB;

  k_zero<<<32, 256, 0, stream>>>(z1, 8192);
  k_zero<<<32, 256, 0, stream>>>(fused, 8192);
  k_cvt_bf16<<<128, 256, 0, stream>>>(Wa, WaB, 131072);
  k_cvt_bf16<<<128, 256, 0, stream>>>(Wb, WbB, 131072);
  if (preconv_wic) {
    k_cvt_bf16<<<32768, 256, 0, stream>>>(W_ic, WicB, 33554432);
    k_img_gemm_bf<<<dim3(16, 8), 256, 0, stream>>>(imgF, WicB, b_ic, img);
  } else {
    k_img_gemm<<<dim3(16, 8), 256, 0, stream>>>(imgF, W_ic, b_ic, img);
  }
  k_attn_gemm<<<dim3(16, 4), 256, 0, stream>>>(img, WaB, ba, WbB, bb, h);
  k_score<<<4, 256, 0, stream>>>(h, Wc, bc, score);
  k_pool<<<16, 256, 0, stream>>>(img, score, pool);
  k_text<<<32, 256, 0, stream>>>(textF, Wt, bt, textr, textB);
  k_dense512_relu<<<32, 256, 0, stream>>>(pool, Wh1, bh1, h1);
  k_z1<<<2048, 256, 0, stream>>>(Wz1, textB, pool, z1);
  k_gate<<<32, 256, 0, stream>>>(z1, bz1, h1, gz);
  k_dense512_relu<<<32, 256, 0, stream>>>(gz, Wo1, bo1, o1);
  k_pmat<<<8192, 256, 0, stream>>>(o1, textr, (unsigned*)P);
  k_fused<<<dim3(32, 16), 256, 0, stream>>>(Wpf, P, fused);
  k_final<<<16, 256, 0, stream>>>(fused, bpf, Wcls, bcls, out);
}
